// Pointnet2Backbone_3822520893468
// MI455X (gfx1250) — matrix operand order repaired and checked by reading, not compiled, not run on hardware
//
#include <hip/hip_runtime.h>
#include <hip/hip_bf16.h>

typedef _Float16 half_t;
typedef __attribute__((ext_vector_type(16))) _Float16 v16h;
typedef __attribute__((ext_vector_type(8)))  _Float16 v8h;
typedef __attribute__((ext_vector_type(8)))  float    v8f;

#define FPS_THREADS 256

// ---------------------------------------------------------------------------
// Furthest point sampling: one block per batch, matches reference semantics
// (dist init +inf, start at index 0, update-min then argmax each step,
//  lowest-index tie break like jnp.argmax).
// ---------------------------------------------------------------------------
__global__ void fps_kernel(const float* __restrict__ xyz, float* __restrict__ temp,
                           int* __restrict__ idxs, int N, int S) {
  int b = blockIdx.x;
  int tid = threadIdx.x;
  const float* p = xyz + (size_t)b * N * 3;
  float* t = temp + (size_t)b * N;
  int* out = idxs + (size_t)b * S;
  __shared__ float sd[FPS_THREADS];
  __shared__ int   si[FPS_THREADS];
  for (int k = tid; k < N; k += FPS_THREADS) t[k] = 1e10f;
  if (tid == 0) out[0] = 0;
  __syncthreads();
  int old = 0;
  for (int j = 1; j < S; ++j) {
    float x1 = p[old * 3 + 0], y1 = p[old * 3 + 1], z1 = p[old * 3 + 2];
    float bestd = -1.f; int besti = 0;
    for (int k = tid; k < N; k += FPS_THREADS) {
      float dx = p[k * 3 + 0] - x1, dy = p[k * 3 + 1] - y1, dz = p[k * 3 + 2] - z1;
      float d = dx * dx + dy * dy + dz * dz;
      float d2 = fminf(d, t[k]);
      t[k] = d2;
      if (d2 > bestd) { bestd = d2; besti = k; }
    }
    sd[tid] = bestd; si[tid] = besti;
    __syncthreads();
    for (int off = FPS_THREADS / 2; off > 0; off >>= 1) {
      if (tid < off) {
        float dv = sd[tid + off]; int iv = si[tid + off];
        if (dv > sd[tid] || (dv == sd[tid] && iv < si[tid])) { sd[tid] = dv; si[tid] = iv; }
      }
      __syncthreads();
    }
    old = si[0];
    if (tid == 0) out[j] = old;
    __syncthreads();
  }
}

__global__ void gather_newxyz(const float* __restrict__ xyz, const int* __restrict__ idxs,
                              float* __restrict__ nx, int N, int S, int total) {
  int i = blockIdx.x * blockDim.x + threadIdx.x;
  if (i >= total) return;                 // total = B*S
  int b = i / S;
  int n = idxs[i];
  nx[i * 3 + 0] = xyz[((size_t)b * N + n) * 3 + 0];
  nx[i * 3 + 1] = xyz[((size_t)b * N + n) * 3 + 1];
  nx[i * 3 + 2] = xyz[((size_t)b * N + n) * 3 + 2];
}

// ---------------------------------------------------------------------------
// Ball query: first-in-index-order up-to-ns hits within radius, pad w/ first.
// ---------------------------------------------------------------------------
__global__ void ball_query(const float* __restrict__ xyz, const float* __restrict__ nx,
                           int* __restrict__ grp, int N, int S, int ns, float r2, int total) {
  int i = blockIdx.x * blockDim.x + threadIdx.x;
  if (i >= total) return;                 // total = B*S
  int b = i / S;
  const float* p = xyz + (size_t)b * N * 3;
  float cx = nx[i * 3 + 0], cy = nx[i * 3 + 1], cz = nx[i * 3 + 2];
  int* g = grp + (size_t)i * ns;
  int cnt = 0, first = N - 1;             // N-1 matches reference's no-hit edge case
  for (int n = 0; n < N; ++n) {
    float dx = p[n * 3 + 0] - cx, dy = p[n * 3 + 1] - cy, dz = p[n * 3 + 2] - cz;
    if (dx * dx + dy * dy + dz * dz < r2) {
      if (cnt == 0) first = n;
      g[cnt++] = n;
      if (cnt == ns) break;
    }
  }
  for (int k = cnt; k < ns; ++k) g[k] = first;
}

// ---------------------------------------------------------------------------
// Convert fp32 weights [Cout, Cin] -> zero-padded fp16 [Cout, Kpad], once per
// layer per stage (reused by all batches). Removes all guards/converts from
// the GEMM hot loop.
// ---------------------------------------------------------------------------
__global__ void convert_w(const float* __restrict__ W, half_t* __restrict__ Wh,
                          int Cin, int Kpad, int total) {
  int i = blockIdx.x * blockDim.x + threadIdx.x;
  if (i >= total) return;                 // total = Cout*Kpad
  int k = i % Kpad; int c = i / Kpad;
  Wh[i] = (k < Cin) ? (half_t)W[(size_t)c * Cin + k] : (half_t)0.f;
}

// ---------------------------------------------------------------------------
// Build grouped feature matrix X [S*ns, Kpad] in fp16 for one batch:
// cols 0..2 = gxyz - center, cols 3..3+C-1 = gathered feats, rest zero pad.
// ---------------------------------------------------------------------------
__global__ void build_x(const float* __restrict__ xyz, const float* __restrict__ nx,
                        const float* __restrict__ feats, const int* __restrict__ grp,
                        half_t* __restrict__ X, int b, int N, int S, int ns, int C,
                        int Kpad, int total) {
  int i = blockIdx.x * blockDim.x + threadIdx.x;
  if (i >= total) return;                 // total = S*ns*Kpad
  int kk = i % Kpad; int row = i / Kpad;
  int j = row % ns;  int s = row / ns;
  int n = grp[((size_t)b * S + s) * ns + j];
  float v = 0.f;
  if (kk < 3)          v = xyz[((size_t)b * N + n) * 3 + kk] - nx[((size_t)b * S + s) * 3 + kk];
  else if (kk < 3 + C) v = feats[((size_t)b * N + n) * C + (kk - 3)];
  X[(size_t)row * Kpad + kk] = (half_t)v;
}

// ---------------------------------------------------------------------------
// WMMA GEMM + fused bias/BN/ReLU epilogue. One 16x16 output tile per wave,
// 4 waves per block, KPAD compile-time so the K loop fully unrolls.
// A layout (16-bit 16x32): lane = row (mod 16), halves give K groups of 8.
// B layout (16-bit 32x16): lane = col (mod 16), halves give K groups of 8, same order as A.
// Final layer fuses the neighborhood max-pool via cross-half shuffle +
// integer atomicMax on fp32 bits (valid: post-ReLU values >= 0).
// ---------------------------------------------------------------------------
template <int KPAD>
__global__ void __launch_bounds__(128) wmma_mlp(
    const half_t* __restrict__ X, const half_t* __restrict__ Wh,
    const float* __restrict__ bias, const float* __restrict__ gamma,
    const float* __restrict__ beta,
    half_t* __restrict__ Y, float* __restrict__ Pool,
    int M, int Cout, int nsLog2, int finalLayer) {
  int wave = threadIdx.x >> 5;
  int lane = threadIdx.x & 31;
  int tileM = blockIdx.x * 4 + wave;
  if (tileM * 16 >= M) return;            // whole-wave exit, EXEC stays all-ones
  int tileN = blockIdx.y;
  int hi = lane >> 4;                     // 0 or 1 (half of wave)
  int l15 = lane & 15;
  int rowA = tileM * 16 + l15;
  int colB = tileN * 16 + l15;
  const half_t* xrow = X + (size_t)rowA * KPAD + hi * 8;
  const half_t* wrow = Wh + (size_t)colB * KPAD + hi * 8;

  v8f acc = {};
#pragma unroll
  for (int k0 = 0; k0 < KPAD; k0 += 32) {
    v8h alo = *(const v8h*)(xrow + k0);       // 16B aligned
    v8h ahi = *(const v8h*)(xrow + k0 + 16);
    v8h blo = *(const v8h*)(wrow + k0);       // 16B aligned (hi*8 halves = 16B)
    v8h bhi = *(const v8h*)(wrow + k0 + 16);
    v16h a = __builtin_shufflevector(alo, ahi, 0, 1, 2, 3, 4, 5, 6, 7,
                                     8, 9, 10, 11, 12, 13, 14, 15);
    v16h bm = __builtin_shufflevector(blo, bhi, 0, 1, 2, 3, 4, 5, 6, 7,
                                      8, 9, 10, 11, 12, 13, 14, 15);
    acc = __builtin_amdgcn_wmma_f32_16x16x32_f16(false, a, false, bm,
                                                 (short)0, acc, false, false);
  }

  float g = gamma[colB], bb = bias[colB], be = beta[colB];
  if (!finalLayer) {
#pragma unroll
    for (int r = 0; r < 8; ++r) {
      int row = tileM * 16 + hi * 8 + r;  // C/D layout: VGPR r, halves = rows +0/+8
      float y = fmaxf(g * (acc[r] + bb) + be, 0.f);
      Y[(size_t)row * Cout + colB] = (half_t)y;   // stride Cout == next layer's KPAD
    }
  } else {
    float vmax = 0.f;                     // ReLU baseline folded into the max
#pragma unroll
    for (int r = 0; r < 8; ++r) vmax = fmaxf(vmax, g * (acc[r] + bb) + be);
    float other = __shfl_xor(vmax, 16, 32);       // merge rows 0-7 with 8-15
    vmax = fmaxf(vmax, other);
    if (hi == 0) {
      int s = (tileM * 16) >> nsLog2;     // ns power of two, tile within one group
      atomicMax((int*)(Pool + (size_t)s * Cout + colB), __float_as_int(vmax));
    }
  }
}

__global__ void transpose_out(const float* __restrict__ f, float* __restrict__ out,
                              int S, int C, int total) {
  int i = blockIdx.x * blockDim.x + threadIdx.x;
  if (i >= total) return;                 // out[b,c,s] = f[b,s,c]
  int s = i % S; int c = (i / S) % C; int b = i / (S * C);
  out[i] = f[((size_t)b * S + s) * C + c];
}

static void launch_mlp(int Kp, const half_t* X, const half_t* Wh,
                       const float* bias, const float* gamma, const float* beta,
                       half_t* Y, float* Pool, int M, int Cout, int nsLog2,
                       int finalLayer, hipStream_t stream) {
  dim3 grid((M / 16 + 3) / 4, Cout / 16);
  switch (Kp) {
    case 32:  hipLaunchKernelGGL(wmma_mlp<32>,  grid, 128, 0, stream, X, Wh, bias, gamma, beta, Y, Pool, M, Cout, nsLog2, finalLayer); break;
    case 64:  hipLaunchKernelGGL(wmma_mlp<64>,  grid, 128, 0, stream, X, Wh, bias, gamma, beta, Y, Pool, M, Cout, nsLog2, finalLayer); break;
    case 128: hipLaunchKernelGGL(wmma_mlp<128>, grid, 128, 0, stream, X, Wh, bias, gamma, beta, Y, Pool, M, Cout, nsLog2, finalLayer); break;
    case 160: hipLaunchKernelGGL(wmma_mlp<160>, grid, 128, 0, stream, X, Wh, bias, gamma, beta, Y, Pool, M, Cout, nsLog2, finalLayer); break;
    default:  hipLaunchKernelGGL(wmma_mlp<288>, grid, 128, 0, stream, X, Wh, bias, gamma, beta, Y, Pool, M, Cout, nsLog2, finalLayer); break;
  }
}

// ---------------------------------------------------------------------------
extern "C" void kernel_launch(void* const* d_in, const int* in_sizes, int n_in,
                              void* d_out, int out_size, void* d_ws, size_t ws_size,
                              hipStream_t stream) {
  (void)in_sizes; (void)n_in; (void)out_size; (void)ws_size;
  const float* pc = (const float*)d_in[0];
  const int B = 8;

  char* ws = (char*)d_ws;
  size_t off = 0;
  auto alloc = [&](size_t bytes) -> char* {
    char* p = ws + off;
    off += (bytes + 255) & ~(size_t)255;
    return p;
  };

  const int  Ss[4]   = {2048, 1024, 512, 256};
  const int  nss[4]  = {64, 32, 16, 16};
  const int  nsl2[4] = {6, 5, 4, 4};
  const float rad[4] = {0.2f, 0.4f, 0.8f, 1.2f};
  const int dims[4][4] = {{3, 64, 64, 128},
                          {131, 128, 128, 256},
                          {259, 128, 128, 256},
                          {259, 128, 128, 128}};

  float* fpstemp = (float*)alloc((size_t)B * 16384 * 4);
  float* newxyz[4]; int* fpsidx[4]; int* grp[4]; float* feats[4];
  for (int s = 0; s < 4; ++s) {
    newxyz[s] = (float*)alloc((size_t)B * Ss[s] * 3 * 4);
    fpsidx[s] = (int*)alloc((size_t)B * Ss[s] * 4);
    grp[s]    = (int*)alloc((size_t)B * Ss[s] * nss[s] * 4);
    feats[s]  = (float*)alloc((size_t)B * Ss[s] * dims[s][3] * 4);
  }
  // fp16 weight buffers (zero-padded to Kpad), one per layer, reused per stage
  half_t* whb[3];
  for (int l = 0; l < 3; ++l) whb[l] = (half_t*)alloc((size_t)256 * 288 * 2);
  // ping-pong fp16 GEMM buffers, sized for the per-batch max (stage1: 131072x64)
  const size_t pingElems = (size_t)2048 * 64 * 64;
  half_t* bufA = (half_t*)alloc(pingElems * 2);
  half_t* bufB = (half_t*)alloc(pingElems * 2);

  const float* cur_xyz = pc;
  const float* cur_feats = nullptr;
  int curN = 16384, curC = 0;

  for (int st = 0; st < 4; ++st) {
    int S = Ss[st], ns = nss[st];
    float r2 = rad[st] * rad[st];
    int totBS = B * S;

    fps_kernel<<<B, FPS_THREADS, 0, stream>>>(cur_xyz, fpstemp, fpsidx[st], curN, S);
    gather_newxyz<<<(totBS + 255) / 256, 256, 0, stream>>>(cur_xyz, fpsidx[st], newxyz[st],
                                                           curN, S, totBS);
    ball_query<<<(totBS + 255) / 256, 256, 0, stream>>>(cur_xyz, newxyz[st], grp[st],
                                                        curN, S, ns, r2, totBS);
    hipMemsetAsync(feats[st], 0, (size_t)B * S * dims[st][3] * 4, stream);

    int Kpad0 = (dims[st][0] + 31) & ~31;
    int M = S * ns;
    int pidx = 1 + st * 12;               // d_in: pointcloud, then 4 stages x 3 x (W,b,g,be)

    // Pre-convert this stage's weights to padded fp16 (shared by all batches)
    int Kps[3];
    for (int l = 0; l < 3; ++l) {
      int Cin = dims[st][l], Cout = dims[st][l + 1];
      Kps[l] = (l == 0) ? Kpad0 : Cin;    // intermediate Couts are multiples of 32
      int totW = Cout * Kps[l];
      convert_w<<<(totW + 255) / 256, 256, 0, stream>>>(
          (const float*)d_in[pidx + l * 4 + 0], whb[l], Cin, Kps[l], totW);
    }

    for (int b = 0; b < B; ++b) {
      int totX = M * Kpad0;
      build_x<<<(totX + 255) / 256, 256, 0, stream>>>(cur_xyz, newxyz[st], cur_feats, grp[st],
                                                      bufA, b, curN, S, ns, curC, Kpad0, totX);
      half_t* xin = bufA; half_t* yout = bufB;
      for (int l = 0; l < 3; ++l) {
        int Cout = dims[st][l + 1];
        const float* bb = (const float*)d_in[pidx + l * 4 + 1];
        const float* gg = (const float*)d_in[pidx + l * 4 + 2];
        const float* be = (const float*)d_in[pidx + l * 4 + 3];
        int finalL = (l == 2);
        launch_mlp(Kps[l], xin, whb[l], bb, gg, be,
                   finalL ? nullptr : yout,
                   finalL ? (feats[st] + (size_t)b * S * Cout) : nullptr,
                   M, Cout, nsl2[st], finalL, stream);
        if (!finalL) { half_t* t = xin; xin = yout; yout = t; }
      }
    }
    cur_xyz = newxyz[st]; cur_feats = feats[st]; curN = S; curC = dims[st][3];
  }

  int total = B * 128 * 256;
  transpose_out<<<(total + 255) / 256, 256, 0, stream>>>(feats[3], (float*)d_out,
                                                         256, 128, total);
}
